// SelfAttention_58342835749157
// MI455X (gfx1250) — matrix operand order repaired and checked by reading, not compiled, not run on hardware
//
#include <hip/hip_runtime.h>
#include <hip/hip_fp16.h>
#include <math.h>

typedef __attribute__((ext_vector_type(16))) _Float16 v16h;
typedef __attribute__((ext_vector_type(8)))  _Float16 v8h;
typedef __attribute__((ext_vector_type(8)))  float    v8f;

#define B_   2
#define C_   256
#define N_   2304     // 48*48
#define NH   8
#define DH   32
#define G_   32
#define CG   8        // C_/G_
#define EPS_ 1e-5f
#define SCALE_ 0.0625f  // 1/sqrt(256)

// ---------------- fragment helpers (CDNA5 WMMA 16x16x32 f16 layouts) ----------

__device__ __forceinline__ v16h cat16(v8h lo, v8h hi) {
  return __builtin_shufflevector(lo, hi, 0,1,2,3,4,5,6,7,8,9,10,11,12,13,14,15);
}

// A fragment: 16(M) x 32(K) f16, source row-major with row stride ld (halves).
// Lane l holds row l&15, K = {b8..b8+7, b8+16..b8+23}, b8 = 8*(l>=16).
__device__ __forceinline__ v16h load_a_frag(const _Float16* base, int ld, int lane) {
  const _Float16* p = base + (size_t)(lane & 15) * ld + ((lane >> 4) << 3);
  v8h lo = *(const v8h*)(p);
  v8h hi = *(const v8h*)(p + 16);
  return cat16(lo, hi);
}

// B fragment: 32(K) x 16(cols) f16, source K-major per column, column stride ldc (halves).
// Lane l holds col l&15, K = {b8..b8+7, b8+16..b8+23}, b8 = 8*(l>=16).
__device__ __forceinline__ v16h load_b_frag(const _Float16* base, int ldc, int lane) {
  const _Float16* p = base + (size_t)(lane & 15) * ldc + ((lane >> 4) << 3);
  v8h lo = *(const v8h*)(p);
  v8h hi = *(const v8h*)(p + 16);
  return cat16(lo, hi);
}

__device__ __forceinline__ v8f wmma16(v16h a, v16h b, v8f c) {
  return __builtin_amdgcn_wmma_f32_16x16x32_f16(false, a, false, b, (short)0, c, false, false);
}

// ---------------- CDNA5 async global->LDS copy (ASYNCcnt path) ----------------

__device__ __forceinline__ void async_b128(unsigned lds_addr, const void* gaddr) {
  asm volatile("global_load_async_to_lds_b128 %0, %1, off"
               :: "v"(lds_addr), "v"(gaddr) : "memory");
}
__device__ __forceinline__ void wait_async0() {
  asm volatile("s_wait_asynccnt 0x0" ::: "memory");
}

// Stage one 32-KV chunk: K tile (32 rows x 32 halves) + V tile (32 rows x 32 halves).
__device__ __forceinline__ void stage_kv(const _Float16* kbase, const _Float16* vbase,
                                         int j0, unsigned lk, unsigned lv, int lane) {
  const char* kg = (const char*)(kbase + (size_t)(j0 + lane) * DH);  // k row = kv index
  const char* vg = (const char*)(vbase + (size_t)lane * N_ + j0);    // v row = d index
  unsigned lkr = lk + (unsigned)lane * 64;
  unsigned lvr = lv + (unsigned)lane * 64;
#pragma unroll
  for (int s = 0; s < 4; ++s) {
    async_b128(lkr + 16u * s, kg + 16 * s);
    async_b128(lvr + 16u * s, vg + 16 * s);
  }
}

// ---------------- kernel 1: GroupNorm -> f16 tokens transposed (B, N, C) -----

__global__ void gn_kernel(const float* __restrict__ x, const float* __restrict__ gw,
                          const float* __restrict__ gb, _Float16* __restrict__ tokT) {
  const int b   = blockIdx.x >> 5;
  const int g   = blockIdx.x & 31;
  const int tid = threadIdx.x;
  const int cnt = CG * N_;                      // 18432 elems per group
  const float* xb = x + ((size_t)b * C_ + g * CG) * N_;

  float s = 0.f, s2 = 0.f;
  for (int i = tid; i < cnt; i += 256) {
    float v = xb[i];
    s += v; s2 += v * v;
  }
  __shared__ float sh[512];
  sh[tid] = s; sh[tid + 256] = s2;
  __syncthreads();
  for (int off = 128; off > 0; off >>= 1) {
    if (tid < off) { sh[tid] += sh[tid + off]; sh[tid + 256] += sh[tid + 256 + off]; }
    __syncthreads();
  }
  const float mean = sh[0] * (1.0f / cnt);
  const float var  = sh[256] * (1.0f / cnt) - mean * mean;
  const float rstd = rsqrtf(var + EPS_);

  float wv[CG], bv[CG];
#pragma unroll
  for (int cc = 0; cc < CG; ++cc) {
    wv[cc] = gw[g * CG + cc] * rstd;
    bv[cc] = gb[g * CG + cc];
  }
  // one token (8 channels) per thread -> one 16B vector store
  for (int n = tid; n < N_; n += 256) {
    v8h ov;
#pragma unroll
    for (int cc = 0; cc < CG; ++cc) {
      ov[cc] = (_Float16)((xb[(size_t)cc * N_ + n] - mean) * wv[cc] + bv[cc]);
    }
    *(v8h*)&tokT[((size_t)b * N_ + n) * C_ + g * CG] = ov;
  }
}

// ---------------- kernel 2: fp32 -> f16 weight conversion ---------------------

__global__ void wcvt_kernel(const float* __restrict__ wq, const float* __restrict__ wk,
                            const float* __restrict__ wv, const float* __restrict__ wo,
                            _Float16* __restrict__ q16, _Float16* __restrict__ k16,
                            _Float16* __restrict__ v16, _Float16* __restrict__ o16) {
  int i = blockIdx.x * 256 + threadIdx.x;
  if (i < C_ * C_) {
    q16[i] = (_Float16)wq[i];
    k16[i] = (_Float16)wk[i];
    v16[i] = (_Float16)wv[i];
    o16[i] = (_Float16)wo[i];
  }
}

// ---------------- kernel 3: fused QKV projection ------------------------------
// q,k -> (b, h, N, d) ; v -> (b, h, d, N) (transposed for P*V B-operand)

__global__ void qkv_kernel(const _Float16* __restrict__ tokT,
                           const _Float16* __restrict__ wq, const _Float16* __restrict__ wk,
                           const _Float16* __restrict__ wv,
                           _Float16* __restrict__ q, _Float16* __restrict__ k,
                           _Float16* __restrict__ vT) {
  const int n0   = blockIdx.x << 4;
  const int m0   = blockIdx.y << 4;
  const int b    = blockIdx.z;
  const int lane = threadIdx.x;

  const _Float16* tb = tokT + ((size_t)b * N_ + n0) * C_;
  v8f aq = {}, ak = {}, av = {};
#pragma unroll 4
  for (int c0 = 0; c0 < C_; c0 += 32) {
    v16h bt = load_b_frag(tb + c0, C_, lane);
    v16h a;
    a  = load_a_frag(wq + (size_t)m0 * C_ + c0, C_, lane);
    aq = wmma16(a, bt, aq);
    a  = load_a_frag(wk + (size_t)m0 * C_ + c0, C_, lane);
    ak = wmma16(a, bt, ak);
    a  = load_a_frag(wv + (size_t)m0 * C_ + c0, C_, lane);
    av = wmma16(a, bt, av);
  }
  const int mbase = (lane >> 4) << 3;  // 0 or 8
  const int n     = n0 + (lane & 15);
  const int hh    = m0 >> 5;
#pragma unroll
  for (int r = 0; r < 8; ++r) {
    int o  = m0 + mbase + r;
    int dd = o & 31;
    size_t qi = (((size_t)b * NH + hh) * N_ + n) * DH + dd;
    q[qi] = (_Float16)aq[r];
    k[qi] = (_Float16)ak[r];
    vT[(((size_t)b * NH + hh) * DH + dd) * N_ + n] = (_Float16)av[r];
  }
}

// ---------------- kernel 4: flash attention, async double-buffered K/V --------

__global__ void attn_kernel(const _Float16* __restrict__ q, const _Float16* __restrict__ k,
                            const _Float16* __restrict__ vT, _Float16* __restrict__ outT) {
  const int n0   = blockIdx.x << 4;
  const int hh   = blockIdx.y;
  const int b    = blockIdx.z;
  const int lane = threadIdx.x;

  __shared__ _Float16 kvbuf[2][2048];   // [buf]: K tile (1024 halves) + V tile (1024 halves)
  __shared__ _Float16 pbuf[16 * 32];

  const size_t bh = (size_t)b * NH + hh;
  const v16h aq = load_a_frag(q + (bh * N_ + n0) * DH, DH, lane);  // q tile reused all iters
  const _Float16* kbase = k  + bh * N_ * DH;
  const _Float16* vbase = vT + bh * DH * N_;
  const unsigned lds0 = (unsigned)(size_t)(const void*)&kvbuf[0][0];

  v8f o0 = {}, o1 = {};
  float mrow[8], lrow[8];
#pragma unroll
  for (int r = 0; r < 8; ++r) { mrow[r] = -1e30f; lrow[r] = 0.f; }

  // prologue: stage first chunk into buffer 0
  stage_kv(kbase, vbase, 0, lds0, lds0 + 2048, lane);

  for (int j0 = 0; j0 < N_; j0 += 32) {
    const int buf = (j0 >> 5) & 1;
    const _Float16* lk = &kvbuf[buf][0];
    const _Float16* lv = &kvbuf[buf][1024];

    wait_async0();  // current buffer resident in LDS
    if (j0 + 32 < N_) {
      const unsigned nb = lds0 + (unsigned)(buf ^ 1) * 4096;
      stage_kv(kbase, vbase, j0 + 32, nb, nb + 2048, lane);  // prefetch rides under compute
    }

    v16h kb0 = load_b_frag(lk, 32, lane);
    v16h kb1 = load_b_frag(lk + 16 * 32, 32, lane);
    v8f s0 = {}, s1 = {};
    s0 = wmma16(aq, kb0, s0);
    s1 = wmma16(aq, kb1, s1);

    // online softmax; row m lives across one 16-lane half (masks 1/2/4/8 stay inside)
#pragma unroll
    for (int r = 0; r < 8; ++r) {
      float x0 = s0[r] * SCALE_;
      float x1 = s1[r] * SCALE_;
      float cm = fmaxf(x0, x1);
      cm = fmaxf(cm, __shfl_xor(cm, 1, 32));
      cm = fmaxf(cm, __shfl_xor(cm, 2, 32));
      cm = fmaxf(cm, __shfl_xor(cm, 4, 32));
      cm = fmaxf(cm, __shfl_xor(cm, 8, 32));
      float mnew = fmaxf(mrow[r], cm);
      float cr   = __expf(mrow[r] - mnew);
      float p0   = __expf(x0 - mnew);
      float p1   = __expf(x1 - mnew);
      float ps   = p0 + p1;
      ps += __shfl_xor(ps, 1, 32);
      ps += __shfl_xor(ps, 2, 32);
      ps += __shfl_xor(ps, 4, 32);
      ps += __shfl_xor(ps, 8, 32);
      lrow[r] = lrow[r] * cr + ps;
      mrow[r] = mnew;
      o0[r] *= cr;
      o1[r] *= cr;
      // stage P (C/D layout -> LDS row-major) for the A-fragment reload
      int mm = r + ((lane >> 4) << 3);
      int cc = lane & 15;
      pbuf[mm * 32 + cc]      = (_Float16)p0;
      pbuf[mm * 32 + cc + 16] = (_Float16)p1;
    }
    __syncthreads();
    const _Float16* pp = &pbuf[(lane & 15) * 32 + ((lane >> 4) << 3)];
    v16h pa  = cat16(*(const v8h*)pp, *(const v8h*)(pp + 16));
    v16h vb0 = load_b_frag(lv, 32, lane);            // d cols 0..15
    v16h vb1 = load_b_frag(lv + 16 * 32, 32, lane);  // d cols 16..31
    o0 = wmma16(pa, vb0, o0);
    o1 = wmma16(pa, vb1, o1);
    __syncthreads();
  }

#pragma unroll
  for (int r = 0; r < 8; ++r) {
    float inv = 1.0f / lrow[r];
    int mm = r + ((lane >> 4) << 3);
    size_t row = ((size_t)b * N_ + n0 + mm) * C_ + hh * DH;
    outT[row + (lane & 15)]      = (_Float16)(o0[r] * inv);
    outT[row + (lane & 15) + 16] = (_Float16)(o1[r] * inv);
  }
}

// ---------------- kernel 5: out projection + bias + residual ------------------

__global__ void oproj_kernel(const _Float16* __restrict__ aoutT, const _Float16* __restrict__ wo,
                             const float* __restrict__ bo, const float* __restrict__ x,
                             float* __restrict__ out) {
  const int n0   = blockIdx.x << 4;
  const int m0   = blockIdx.y << 4;
  const int b    = blockIdx.z;
  const int lane = threadIdx.x;

  const _Float16* tb = aoutT + ((size_t)b * N_ + n0) * C_;
  v8f acc = {};
#pragma unroll 4
  for (int c0 = 0; c0 < C_; c0 += 32) {
    v16h bt = load_b_frag(tb + c0, C_, lane);
    v16h a  = load_a_frag(wo + (size_t)m0 * C_ + c0, C_, lane);
    acc = wmma16(a, bt, acc);
  }
  const int mbase = (lane >> 4) << 3;
  const int n     = n0 + (lane & 15);
#pragma unroll
  for (int r = 0; r < 8; ++r) {
    int o = m0 + mbase + r;
    size_t idx = ((size_t)b * C_ + o) * N_ + n;
    out[idx] = acc[r] + bo[o] + x[idx];
  }
}

// ---------------- host-side launch --------------------------------------------

extern "C" void kernel_launch(void* const* d_in, const int* in_sizes, int n_in,
                              void* d_out, int out_size, void* d_ws, size_t ws_size,
                              hipStream_t stream) {
  const float* x    = (const float*)d_in[0];
  const float* gn_w = (const float*)d_in[1];
  const float* gn_b = (const float*)d_in[2];
  const float* wq   = (const float*)d_in[3];
  const float* wk   = (const float*)d_in[4];
  const float* wv   = (const float*)d_in[5];
  const float* wo   = (const float*)d_in[6];
  const float* bo   = (const float*)d_in[7];
  float* out = (float*)d_out;

  // workspace carve-up (256B aligned)
  char* ws = (char*)d_ws;
  size_t off = 0;
  auto alloc = [&](size_t bytes) { char* p = ws + off; off = (off + bytes + 255) & ~(size_t)255; return p; };
  const size_t tokB = (size_t)B_ * N_ * C_ * sizeof(_Float16);        // 2.25 MiB
  const size_t wB   = (size_t)C_ * C_ * sizeof(_Float16);             // 128 KiB
  _Float16* tokT  = (_Float16*)alloc(tokB);
  _Float16* wq16  = (_Float16*)alloc(wB);
  _Float16* wk16  = (_Float16*)alloc(wB);
  _Float16* wv16  = (_Float16*)alloc(wB);
  _Float16* wo16  = (_Float16*)alloc(wB);
  _Float16* q16   = (_Float16*)alloc(tokB);
  _Float16* k16   = (_Float16*)alloc(tokB);
  _Float16* vT16  = (_Float16*)alloc(tokB);
  _Float16* aoutT = (_Float16*)alloc(tokB);
  (void)ws_size; (void)in_sizes; (void)n_in; (void)out_size;

  gn_kernel<<<dim3(B_ * G_), dim3(256), 0, stream>>>(x, gn_w, gn_b, tokT);
  wcvt_kernel<<<dim3((C_ * C_ + 255) / 256), dim3(256), 0, stream>>>(
      wq, wk, wv, wo, wq16, wk16, wv16, wo16);
  qkv_kernel<<<dim3(N_ / 16, C_ / 16, B_), dim3(32), 0, stream>>>(
      tokT, wq16, wk16, wv16, q16, k16, vT16);
  attn_kernel<<<dim3(N_ / 16, NH, B_), dim3(32), 0, stream>>>(q16, k16, vT16, aoutT);
  oproj_kernel<<<dim3(N_ / 16, C_ / 16, B_), dim3(32), 0, stream>>>(
      aoutT, wo16, bo, x, out);
}
